// WindowAttention_22101901705472
// MI455X (gfx1250) — compile-verified, matrix operand order checked by reading, not run on hardware
//
#include <hip/hip_runtime.h>
#include <hip/hip_bf16.h>
#include <math.h>

// ---------------- types for WMMA ----------------
typedef __attribute__((ext_vector_type(16))) __bf16 v16bf;
typedef __attribute__((ext_vector_type(8)))  __bf16 v8bf;
typedef __attribute__((ext_vector_type(8)))  float  v8f;

#define NTOK 49
#define NP   64
#define CDIM 128
#define HEADS 4
#define HD   32
#define NWIN 256
#define SCALE 0.17677669529663687f   // 1/sqrt(32)

// LDS layout (bytes), total 160 KB
#define OFF_X    0          // bf16 [64][128] X token-major (overlaid by sO later)
#define OFF_Q    16384      // bf16 [64][128] Q token-major (scaled)
#define OFF_K    32768      // bf16 [64][128] K token-major
#define OFF_VT   49152      // bf16 [128][64] V chan-major (row=channel, col=token)
#define OFF_ST   65536      // f32  [4][64][64] S TRANSPOSED: [h][j][i]
#define OFF_P    131072     // bf16 [4][64][64] P row-major: [h][i][j]
#define SMEM_BYTES 163840

// Load a 16x32 bf16 fragment from a row-major [rows][ld] array (K contiguous).
// A: [M][K]; B: [N][K]. Per ISA 7.12.2: lane&15 -> row, lane>>4 -> K-half.
__device__ inline v16bf load_frag(const __bf16* base, int ld, int lane) {
    int r  = lane & 15;
    int kh = (lane >> 4) << 3;          // 0 or 8
    const __bf16* p = base + r * ld + kh;
    v8bf lo = *(const v8bf*)(p);
    v8bf hi = *(const v8bf*)(p + 16);
    v16bf f;
#pragma unroll
    for (int i = 0; i < 8; ++i) { f[i] = lo[i]; f[i + 8] = hi[i]; }
    return f;
}

__device__ inline v8f wmma_bf16(v16bf a, v16bf b, v8f c) {
    return __builtin_amdgcn_wmma_f32_16x16x32_bf16(false, a, false, b, (short)0, c, false, false);
}

// ---------------- prep: weights -> bf16, gather rel-pos bias ----------------
__global__ void prep_kernel(const float* __restrict__ qkv_w,
                            const float* __restrict__ proj_w,
                            const float* __restrict__ rpb,
                            const int*   __restrict__ rpi,
                            __bf16* __restrict__ gq,
                            __bf16* __restrict__ gp,
                            float*  __restrict__ gbias) {
    int t = blockIdx.x * 256 + threadIdx.x;
    if (t < 384 * 128) gq[t] = (__bf16)qkv_w[t];
    if (t < 128 * 128) gp[t] = (__bf16)proj_w[t];
    if (t < HEADS * NTOK * NTOK) {
        int h  = t / (NTOK * NTOK);
        int ij = t % (NTOK * NTOK);
        gbias[t] = rpb[rpi[ij] * HEADS + h];
    }
}

// ---------------- fused window attention: 1 block = 1 window ----------------
__global__ __launch_bounds__(128)
void win_attn_kernel(const float* __restrict__ x,
                     const float* __restrict__ mask,
                     const float* __restrict__ qkv_b,
                     const float* __restrict__ proj_b,
                     const __bf16* __restrict__ gq,     // [384][128] bf16
                     const __bf16* __restrict__ gp,     // [128][128] bf16
                     const float*  __restrict__ gbias,  // [4][49][49]
                     float* __restrict__ out) {
    extern __shared__ __align__(16) char smem[];
    __bf16* sX  = (__bf16*)(smem + OFF_X);
    __bf16* sQ  = (__bf16*)(smem + OFF_Q);
    __bf16* sK  = (__bf16*)(smem + OFF_K);
    __bf16* sVT = (__bf16*)(smem + OFF_VT);
    float*  sST = (float*)(smem + OFF_ST);
    __bf16* sP  = (__bf16*)(smem + OFF_P);
    __bf16* sO  = (__bf16*)(smem + OFF_X);   // overlays sX (dead after phase A)

    const int b    = blockIdx.x;            // window index in [0, 16384)
    const int g    = b & (NWIN - 1);        // mask group = b % 256
    const int wave = threadIdx.x >> 5;
    const int lane = threadIdx.x & 31;
    const int half = (lane >> 4) << 3;      // 0 or 8

    // prefetch bf16 weights (L2 hint; 131072 B total, 128 threads -> 1 KB each)
    __builtin_prefetch(((const char*)gq) + threadIdx.x * 1024, 0, 1);

    // ---- stage X (fp32 -> bf16, rows 49..63 zero) ----
    for (int i = threadIdx.x; i < NP * 32; i += 128) {
        int row = i >> 5;
        int c4  = (i & 31) << 2;
        float4 v = make_float4(0.f, 0.f, 0.f, 0.f);
        if (row < NTOK) v = *(const float4*)(x + ((size_t)b * NTOK + row) * CDIM + c4);
        __bf16* dst = sX + row * CDIM + c4;
        dst[0] = (__bf16)v.x; dst[1] = (__bf16)v.y;
        dst[2] = (__bf16)v.z; dst[3] = (__bf16)v.w;
    }
    __syncthreads();

    // ---- Phase A1: Q^T = Wq @ X^T  (D rows = channel, cols = token) ----
    // A = gq rows [0,128), B = sX [token][cin]. Store sQ[token][chan] contiguously.
    for (int t = wave; t < 32; t += 4) {
        int mt = t >> 2, nt = t & 3;        // mt: chan tile, nt: token tile
        v8f acc = {};
#pragma unroll
        for (int kk = 0; kk < 4; ++kk) {
            v16bf a  = load_frag(gq + (mt * 16) * CDIM + kk * 32, CDIM, lane);
            v16bf bf = load_frag(sX + (nt * 16) * CDIM + kk * 32, CDIM, lane);
            acc = wmma_bf16(a, bf, acc);
        }
        int tok = nt * 16 + (lane & 15);
        int cb  = mt * 16 + half;           // 8 consecutive channels
        float4 b0 = *(const float4*)(qkv_b + cb);
        float4 b1 = *(const float4*)(qkv_b + cb + 4);
        v8bf o;
        o[0] = (__bf16)((acc[0] + b0.x) * SCALE); o[1] = (__bf16)((acc[1] + b0.y) * SCALE);
        o[2] = (__bf16)((acc[2] + b0.z) * SCALE); o[3] = (__bf16)((acc[3] + b0.w) * SCALE);
        o[4] = (__bf16)((acc[4] + b1.x) * SCALE); o[5] = (__bf16)((acc[5] + b1.y) * SCALE);
        o[6] = (__bf16)((acc[6] + b1.z) * SCALE); o[7] = (__bf16)((acc[7] + b1.w) * SCALE);
        *(v8bf*)(sQ + tok * CDIM + cb) = o;
    }
    // ---- Phase A2: K^T = Wk @ X^T -> sK[token][chan] ----
    for (int t = wave; t < 32; t += 4) {
        int mt = t >> 2, nt = t & 3;
        v8f acc = {};
#pragma unroll
        for (int kk = 0; kk < 4; ++kk) {
            v16bf a  = load_frag(gq + (128 + mt * 16) * CDIM + kk * 32, CDIM, lane);
            v16bf bf = load_frag(sX + (nt * 16) * CDIM + kk * 32, CDIM, lane);
            acc = wmma_bf16(a, bf, acc);
        }
        int tok = nt * 16 + (lane & 15);
        int cb  = mt * 16 + half;
        float4 b0 = *(const float4*)(qkv_b + 128 + cb);
        float4 b1 = *(const float4*)(qkv_b + 128 + cb + 4);
        v8bf o;
        o[0] = (__bf16)(acc[0] + b0.x); o[1] = (__bf16)(acc[1] + b0.y);
        o[2] = (__bf16)(acc[2] + b0.z); o[3] = (__bf16)(acc[3] + b0.w);
        o[4] = (__bf16)(acc[4] + b1.x); o[5] = (__bf16)(acc[5] + b1.y);
        o[6] = (__bf16)(acc[6] + b1.z); o[7] = (__bf16)(acc[7] + b1.w);
        *(v8bf*)(sK + tok * CDIM + cb) = o;
    }
    // ---- Phase A3: V = X @ Wv^T (D rows = token, cols = chan) -> sVT[chan][tok] ----
    for (int t = wave; t < 32; t += 4) {
        int nt = t >> 2, mt = t & 3;        // nt: chan tile, mt: token tile
        v8f acc = {};
#pragma unroll
        for (int kk = 0; kk < 4; ++kk) {
            v16bf a  = load_frag(sX + (mt * 16) * CDIM + kk * 32, CDIM, lane);
            v16bf bf = load_frag(gq + (256 + nt * 16) * CDIM + kk * 32, CDIM, lane);
            acc = wmma_bf16(a, bf, acc);
        }
        int c  = nt * 16 + (lane & 15);
        int tb = mt * 16 + half;            // 8 consecutive tokens
        float bv = qkv_b[256 + c];
        v8bf o;
#pragma unroll
        for (int r = 0; r < 8; ++r) o[r] = (__bf16)(acc[r] + bv);
        *(v8bf*)(sVT + c * NP + tb) = o;
    }
    __syncthreads();

    // ---- Phase B: S = Q @ K^T (per head = per wave), store TRANSPOSED sST[j][i] ----
    const int h = wave;
#pragma unroll
    for (int mt = 0; mt < 4; ++mt) {
#pragma unroll
        for (int nt = 0; nt < 4; ++nt) {
            v16bf a  = load_frag(sQ + (mt * 16) * CDIM + h * HD, CDIM, lane);
            v16bf bf = load_frag(sK + (nt * 16) * CDIM + h * HD, CDIM, lane);
            v8f acc = {};
            acc = wmma_bf16(a, bf, acc);
            int j  = nt * 16 + (lane & 15);
            int ib = mt * 16 + half;        // 8 consecutive i
            float* dst = sST + (h * NP + j) * NP + ib;
            *(float4*)(dst)     = make_float4(acc[0], acc[1], acc[2], acc[3]);
            *(float4*)(dst + 4) = make_float4(acc[4], acc[5], acc[6], acc[7]);
        }
    }
    __syncthreads();

    // ---- softmax over j (fp32) with fused bias+mask; write bf16 P[h][i][j] ----
    const float* biasH = gbias + h * NTOK * NTOK;
    const float* maskG = mask + (size_t)g * NTOK * NTOK;
    for (int row = lane; row < NP; row += 32) {   // row = i (query token)
        __bf16* prow = sP + (h * NP + row) * NP;
        if (row < NTOK) {
            const float* bm = biasH + row * NTOK;
            const float* mk = maskG + row * NTOK;
            float* scol = sST + h * NP * NP + row;   // element j at scol[j*64]
            float mx = -1e30f;
            for (int j = 0; j < NTOK; ++j) {
                float v = scol[j * NP] + bm[j] + mk[j];
                scol[j * NP] = v;
                mx = fmaxf(mx, v);
            }
            float sum = 0.f;
            for (int j = 0; j < NTOK; ++j) {
                float e = __expf(scol[j * NP] - mx);
                sum += e;
                prow[j] = (__bf16)e;
            }
            float rinv = 1.f / sum;
            for (int j = 0; j < NTOK; ++j) prow[j] = (__bf16)((float)prow[j] * rinv);
            for (int j = NTOK; j < NP; ++j) prow[j] = (__bf16)0.f;
        } else {
            for (int j = 0; j < NP; ++j) prow[j] = (__bf16)0.f;
        }
    }
    __syncthreads();

    // ---- Phase C: O^T = V^T @ P^T  (A = sVT[c][j], B = sP[i][j]) ----
    // D rows = channel, cols = token -> store sO[token][chan] contiguously.
#pragma unroll
    for (int ct = 0; ct < 2; ++ct) {
#pragma unroll
        for (int nt = 0; nt < 4; ++nt) {
            v8f acc = {};
#pragma unroll
            for (int kk = 0; kk < 2; ++kk) {
                v16bf a  = load_frag(sVT + (h * HD + ct * 16) * NP + kk * 32, NP, lane);
                v16bf bf = load_frag(sP + (h * NP + nt * 16) * NP + kk * 32, NP, lane);
                acc = wmma_bf16(a, bf, acc);
            }
            int tok = nt * 16 + (lane & 15);
            int cb  = h * HD + ct * 16 + half;   // 8 consecutive channels
            v8bf o;
#pragma unroll
            for (int r = 0; r < 8; ++r) o[r] = (__bf16)acc[r];
            *(v8bf*)(sO + tok * CDIM + cb) = o;
        }
    }
    __syncthreads();

    // ---- Phase D: out^T = Wproj @ O^T  (A = gp[n][c], B = sO[tok][c]) ----
    // D rows = out channel, cols = token -> coalesced float4 global stores.
    for (int t = wave; t < 32; t += 4) {
        int mt = t >> 2, nt = t & 3;        // mt: out-chan tile, nt: token tile
        v8f acc = {};
#pragma unroll
        for (int kk = 0; kk < 4; ++kk) {
            v16bf a  = load_frag(gp + (mt * 16) * CDIM + kk * 32, CDIM, lane);
            v16bf bf = load_frag(sO + (nt * 16) * CDIM + kk * 32, CDIM, lane);
            acc = wmma_bf16(a, bf, acc);
        }
        int tok = nt * 16 + (lane & 15);
        int cb  = mt * 16 + half;
        if (tok < NTOK) {
            float4 p0 = *(const float4*)(proj_b + cb);
            float4 p1 = *(const float4*)(proj_b + cb + 4);
            float* dst = out + ((size_t)b * NTOK + tok) * CDIM + cb;
            *(float4*)(dst)     = make_float4(acc[0] + p0.x, acc[1] + p0.y,
                                              acc[2] + p0.z, acc[3] + p0.w);
            *(float4*)(dst + 4) = make_float4(acc[4] + p1.x, acc[5] + p1.y,
                                              acc[6] + p1.z, acc[7] + p1.w);
        }
    }
}

extern "C" void kernel_launch(void* const* d_in, const int* in_sizes, int n_in,
                              void* d_out, int out_size, void* d_ws, size_t ws_size,
                              hipStream_t stream) {
    (void)in_sizes; (void)n_in; (void)out_size; (void)ws_size;
    const float* x      = (const float*)d_in[0];
    const float* mask   = (const float*)d_in[1];
    const float* qkv_w  = (const float*)d_in[2];
    const float* qkv_b  = (const float*)d_in[3];
    const float* proj_w = (const float*)d_in[4];
    const float* proj_b = (const float*)d_in[5];
    const float* rpb    = (const float*)d_in[6];
    const int*   rpi    = (const int*)d_in[7];
    float* out = (float*)d_out;

    char* ws = (char*)d_ws;
    __bf16* gq    = (__bf16*)(ws);            // 384*128*2 = 98304 B
    __bf16* gp    = (__bf16*)(ws + 98304);    // 128*128*2 = 32768 B
    float*  gbias = (float*)(ws + 131072);    // 4*49*49*4 = 38416 B

    prep_kernel<<<192, 256, 0, stream>>>(qkv_w, proj_w, rpb, rpi, gq, gp, gbias);
    win_attn_kernel<<<256 * 64, 128, SMEM_BYTES, stream>>>(
        x, mask, qkv_b, proj_b, gq, gp, gbias, out);
}
